// Attention_9801115369627
// MI455X (gfx1250) — compile-verified, matrix operand order checked by reading, not run on hardware
//
#include <hip/hip_runtime.h>
#include <hip/hip_fp16.h>
#include <hip/hip_bf16.h>

typedef __attribute__((ext_vector_type(16))) _Float16 v16h;
typedef __attribute__((ext_vector_type(8)))  _Float16 v8h;
typedef __attribute__((ext_vector_type(8)))  float    v8f;
typedef __attribute__((ext_vector_type(4)))  float    v4f;
typedef _Float16 half_t;

// ---- problem constants ----
constexpr int kBatch = 32;
constexpr int kDim   = 384;
constexpr int kRes   = 28;
constexpr int kN     = kRes * kRes;      // 784
constexpr int kHeads = 8;
constexpr int kKD    = 32;               // KEY_DIM
constexpr int kD     = 128;              // RATIO*KEY_DIM
constexpr int kDH    = kD * kHeads;      // 1024
constexpr int kHQKV  = kDH + 2 * kKD * kHeads; // 1536
constexpr int kHper  = kHQKV / kHeads;   // 192 (q:32, k:32, v:128)
constexpr int kLDP   = 40;               // padded LDS row stride (halfs)
constexpr float kEps   = 1e-5f;
constexpr float kScale = 0.17677669529663687f; // 32^-0.5

// ---------------- WMMA helpers (16x16x32 f16 -> f32) ----------------
__device__ __forceinline__ v8f wmma_f16(v16h a, v16h b, v8f c) {
  return __builtin_amdgcn_wmma_f32_16x16x32_f16(false, a, false, b, (short)0, c,
                                                false, false);
}

// Fragment whose K-dim is contiguous at `row`: lane's 16 halfs are two
// contiguous 8-half runs (row + 8*hi, row + 16 + 8*hi)  -> 2x b128 loads.
__device__ __forceinline__ v16h frag_row(const half_t* row, int hi) {
  const v8h lo = *(const v8h*)(row + 8 * hi);
  const v8h up = *(const v8h*)(row + 16 + 8 * hi);
  v16h r;
#pragma unroll
  for (int i = 0; i < 8; ++i) { r[i] = lo[i]; r[i + 8] = up[i]; }
  return r;
}

__device__ __forceinline__ v16h frag_row2(const half_t* p_lo, const half_t* p_up) {
  const v8h lo = *(const v8h*)p_lo;
  const v8h up = *(const v8h*)p_up;
  v16h r;
#pragma unroll
  for (int i = 0; i < 8; ++i) { r[i] = lo[i]; r[i + 8] = up[i]; }
  return r;
}

__device__ __forceinline__ float hswish(float x) {
  float t = fminf(fmaxf(x + 3.f, 0.f), 6.f);
  return x * t * (1.f / 6.f);
}

// Wave-local LDS fence (same-wave DS ops are in-order on CDNA5).
__device__ __forceinline__ void wave_lds_fence() {
  __builtin_amdgcn_wave_barrier();
  asm volatile("s_wait_dscnt 0" ::: "memory");
  __builtin_amdgcn_wave_barrier();
}

// ---------------- stage 0: prep ----------------
__global__ void __launch_bounds__(256)
prep_weights(const float* __restrict__ qw, const float* __restrict__ qg,
             const float* __restrict__ qb, const float* __restrict__ qm,
             const float* __restrict__ qv, const float* __restrict__ pw,
             const float* __restrict__ pg, const float* __restrict__ pb,
             const float* __restrict__ pm, const float* __restrict__ pv,
             half_t* __restrict__ qwh, float* __restrict__ qbias,
             half_t* __restrict__ pwh, float* __restrict__ pbias) {
  int i = blockIdx.x * blockDim.x + threadIdx.x;
  int stride = gridDim.x * blockDim.x;
  for (int t = i; t < kHQKV * kDim; t += stride) {
    int o = t / kDim;
    float s = qg[o] * rsqrtf(qv[o] + kEps);
    qwh[t] = (half_t)(qw[t] * s);
  }
  for (int t = i; t < kDim * kDH; t += stride) {
    int o = t / kDH;
    float s = pg[o] * rsqrtf(pv[o] + kEps);
    pwh[t] = (half_t)(pw[t] * s);
  }
  for (int t = i; t < kHQKV; t += stride) {
    float s = qg[t] * rsqrtf(qv[t] + kEps);
    qbias[t] = qb[t] - qm[t] * s;
  }
  for (int t = i; t < kDim; t += stride) {
    float s = pg[t] * rsqrtf(pv[t] + kEps);
    pbias[t] = pb[t] - pm[t] * s;
  }
}

// x [b][c][n] f32 -> xT [b][n][c] f16 ; bias table gather (symmetric in n,m)
__global__ void __launch_bounds__(256)
prep_x_bias(const float* __restrict__ x, half_t* __restrict__ xT,
            const float* __restrict__ ab, const int* __restrict__ idx,
            float* __restrict__ biasF, int n_off) {
  long long i = (long long)blockIdx.x * blockDim.x + threadIdx.x;
  long long stride = (long long)gridDim.x * blockDim.x;
  const long long nx = (long long)kBatch * kDim * kN;
  for (long long t = i; t < nx; t += stride) {
    int b = (int)(t / (kDim * kN));
    int rem = (int)(t % (kDim * kN));
    int c = rem / kN, n = rem % kN;
    xT[((size_t)b * kN + n) * kDim + c] = (half_t)x[t];
  }
  const long long nb = (long long)kHeads * kN * kN;
  for (long long t = i; t < nb; t += stride) {
    int h = (int)(t / ((long long)kN * kN));
    int p = (int)(t % ((long long)kN * kN));
    biasF[t] = ab[h * n_off + idx[p]];
  }
}

// ---------------- stage 1: QKV GEMM + BN + hardswish (no LDS, L2-streaming) ----
// Writes: qT[b,h,n,kd] (f16), kT[b,h,n,kd] (f16), V[b,h,d,n] (f16)
__device__ __forceinline__ void qkv_epilogue(v8f acc, int o0, int b, int n0,
                                             int col, int hi,
                                             const float* __restrict__ bias,
                                             half_t* __restrict__ qT,
                                             half_t* __restrict__ kT,
                                             half_t* __restrict__ Vb) {
  const int h = o0 / kHper;
  const int i0 = o0 - h * kHper;  // 16-row tile lies in one of q/k/v ranges
  v8h vals;
#pragma unroll
  for (int r = 0; r < 8; ++r)
    vals[r] = (half_t)hswish(acc[r] + bias[o0 + r + 8 * hi]);
  const size_t bh = (size_t)b * kHeads + h;
  if (i0 < 2 * kKD) {  // q or k: transposed store, one b128 per lane
    half_t* dst = (i0 < kKD) ? qT : kT;
    int kd0 = (i0 < kKD ? i0 : i0 - kKD) + 8 * hi;
    *(v8h*)(dst + (bh * kN + n0 + col) * kKD + kd0) = vals;
  } else {
    half_t* vb = Vb + (bh * kD + (i0 - 2 * kKD)) * kN;
#pragma unroll
    for (int r = 0; r < 8; ++r)
      vb[(size_t)(r + 8 * hi) * kN + n0 + col] = vals[r];
  }
}

__global__ void __launch_bounds__(256)
qkv_gemm(const half_t* __restrict__ W, const float* __restrict__ bias,
         const half_t* __restrict__ xT, half_t* __restrict__ qT,
         half_t* __restrict__ kT, half_t* __restrict__ Vb) {
  const int b = blockIdx.z;
  const int n0 = blockIdx.y * 16;
  const int wave = threadIdx.x >> 5;
  const int o0 = (blockIdx.x * 8 + wave) * 32;  // two 16-row output tiles
  const int lane = threadIdx.x & 31;
  const int col = lane & 15, hi = lane >> 4;
  const half_t* xrow = xT + ((size_t)b * kN + n0 + col) * kDim;
  const half_t* wrow0 = W + (size_t)(o0 + col) * kDim;
  const half_t* wrow1 = W + (size_t)(o0 + 16 + col) * kDim;
  v8f acc0 = {}, acc1 = {};
#pragma unroll 4
  for (int k0 = 0; k0 < kDim; k0 += 32) {
    if (k0 + 64 < kDim) __builtin_prefetch(xrow + k0 + 64, 0, 3);
    v16h bf = frag_row(xrow + k0, hi);
    v16h a0 = frag_row(wrow0 + k0, hi);
    v16h a1 = frag_row(wrow1 + k0, hi);
    acc0 = wmma_f16(a0, bf, acc0);
    acc1 = wmma_f16(a1, bf, acc1);
  }
  qkv_epilogue(acc0, o0, b, n0, col, hi, bias, qT, kT, Vb);
  qkv_epilogue(acc1, o0 + 16, b, n0, col, hi, bias, qT, kT, Vb);
}

// ---------------- stage 2: flash attention per (b, h, 16-query tile) ----------------
__global__ void __launch_bounds__(128)
attn_kernel(const half_t* __restrict__ qT, const half_t* __restrict__ kT,
            const half_t* __restrict__ Vb, const float* __restrict__ biasF,
            half_t* __restrict__ hT) {
  __shared__ alignas(16) half_t lds_p[4][16 * kLDP];
  __shared__ float lds_f[4][16];
  const int wave = threadIdx.x >> 5;
  const int lane = threadIdx.x & 31;
  const int col = lane & 15, hi = lane >> 4;
  const int ntile = blockIdx.x * 4 + wave;
  if (ntile >= 49) return;  // no block-wide syncs below
  const int b = blockIdx.z, h = blockIdx.y;
  const int n0 = ntile * 16;
  const size_t bh = (size_t)b * kHeads + h;

  const half_t* qrow = qT + (bh * kN) * kKD;
  const half_t* krow = kT + (bh * kN) * kKD;
  const half_t* vy = Vb + bh * kD * kN;
  const float* bhp = biasF + (size_t)h * kN * kN;
  half_t* lp = lds_p[wave];
  float* lf = lds_f[wave];

  // A = Q^T tile [16 n x 32 kd]: per-lane 2x b128 from qT
  v16h aq = frag_row(qrow + (size_t)(n0 + col) * kKD, hi);

  v8f O[8];
#pragma unroll
  for (int t = 0; t < 8; ++t) O[t] = (v8f){};
  float rowm[8], rowl[8];
#pragma unroll
  for (int r = 0; r < 8; ++r) { rowm[r] = -1e30f; rowl[r] = 0.f; }

  for (int mt = 0; mt < 25; ++mt) {
    const int m0 = mt * 32;
    const bool have2 = (m0 + 16) < kN;  // last tile covers only 16 keys

    // S tiles: B operand from kT (kd contiguous per key) -> 2x b128 each
    v16h bk = frag_row(krow + (size_t)(m0 + col) * kKD, hi);
    v8f z = {};
    v8f s1 = wmma_f16(aq, bk, z);
    v8f s2 = z;
    if (have2) {
      v16h bk2 = frag_row(krow + (size_t)(m0 + 16 + col) * kKD, hi);
      s2 = wmma_f16(aq, bk2, z);
    }

    // bias (symmetric table read transposed -> vector loads) + scale
    float sv1[8], sv2[8], alpha_r[8];
    {
      const float* bc1 = bhp + (size_t)(m0 + col) * kN + n0 + 8 * hi;
      v4f b10 = *(const v4f*)bc1;
      v4f b11 = *(const v4f*)(bc1 + 4);
      v4f b20 = {}, b21 = {};
      if (have2) {
        const float* bc2 = bhp + (size_t)(m0 + 16 + col) * kN + n0 + 8 * hi;
        b20 = *(const v4f*)bc2;
        b21 = *(const v4f*)(bc2 + 4);
      }
#pragma unroll
      for (int r = 0; r < 8; ++r) {
        float bias1 = (r < 4) ? b10[r & 3] : b11[r & 3];
        float bias2 = (r < 4) ? b20[r & 3] : b21[r & 3];
        sv1[r] = s1[r] * kScale + bias1;
        sv2[r] = have2 ? (s2[r] * kScale + bias2) : -1e30f;
      }
    }

    // online softmax (row stats via wave32 xor-shuffles within 16-lane group)
#pragma unroll
    for (int r = 0; r < 8; ++r) {
      float mx = fmaxf(sv1[r], sv2[r]);
#pragma unroll
      for (int xm = 1; xm < 16; xm <<= 1) mx = fmaxf(mx, __shfl_xor(mx, xm, 32));
      float mnew = fmaxf(rowm[r], mx);
      alpha_r[r] = __expf(rowm[r] - mnew);
      rowm[r] = mnew;
      float p1 = __expf(sv1[r] - mnew);
      float p2 = have2 ? __expf(sv2[r] - mnew) : 0.f;
      float rs = p1 + p2;
#pragma unroll
      for (int xm = 1; xm < 16; xm <<= 1) rs += __shfl_xor(rs, xm, 32);
      rowl[r] = rowl[r] * alpha_r[r] + rs;
      sv1[r] = p1;
      sv2[r] = p2;
    }

    // P (C-layout) -> LDS [n][mm]; alpha broadcast per n
#pragma unroll
    for (int r = 0; r < 8; ++r) {
      int nrow = r + 8 * hi;
      lp[nrow * kLDP + col] = (half_t)sv1[r];
      lp[nrow * kLDP + 16 + col] = (half_t)sv2[r];
    }
    if (col < 8) lf[col + 8 * hi] = alpha_r[col];
    wave_lds_fence();

    float al = lf[col];  // alpha for my output column n
#pragma unroll
    for (int t = 0; t < 8; ++t)
#pragma unroll
      for (int r = 0; r < 8; ++r) O[t][r] *= al;

    // B = P^T [mm x n] from LDS: 2x ds_load_b128
    v16h pb = frag_row(lp + col * kLDP, hi);
    wave_lds_fence();

    // O[d, n] += V[d, mm] * P^T[mm, n]; A = V rows (mm contiguous) -> 2x b128
#pragma unroll
    for (int t = 0; t < 8; ++t) {
      const half_t* vrow = vy + (size_t)(t * 16 + col) * kN + m0;
      const half_t* p_lo = vrow + 8 * hi;
      const half_t* p_up = have2 ? (vrow + 16 + 8 * hi) : p_lo;  // P==0 there
      v16h av = frag_row2(p_lo, p_up);
      O[t] = wmma_f16(av, pb, O[t]);
    }
  }

  // normalize + hardswish + store transposed hT[b, n, cd] (contiguous v8h)
  if (col < 8) lf[col + 8 * hi] = rowl[col];
  wave_lds_fence();
  float linv = 1.f / lf[col];
  half_t* hrow = hT + ((size_t)b * kN + n0 + col) * kDH + (size_t)h * kD;
#pragma unroll
  for (int t = 0; t < 8; ++t) {
    v8h hv;
#pragma unroll
    for (int r = 0; r < 8; ++r) hv[r] = (half_t)hswish(O[t][r] * linv);
    *(v8h*)(hrow + t * 16 + 8 * hi) = hv;
  }
}

// ---------------- stage 3: projection GEMM + BN + hardswish (f32 out) ----------------
__global__ void __launch_bounds__(256)
proj_gemm(const half_t* __restrict__ W, const float* __restrict__ bias,
          const half_t* __restrict__ hT, float* __restrict__ Out) {
  const int b = blockIdx.z;
  const int n0 = blockIdx.y * 32;  // two 16-col tiles (second may be absent)
  const bool have2 = (n0 + 16) < kN;  // kN/16 = 49 tiles (odd)
  const int wave = threadIdx.x >> 5;
  const int o0 = (blockIdx.x * 8 + wave) * 16;
  const int lane = threadIdx.x & 31;
  const int col = lane & 15, hi = lane >> 4;
  const half_t* wrow = W + (size_t)(o0 + col) * kDH;
  const half_t* xrow0 = hT + ((size_t)b * kN + n0 + col) * kDH;
  const half_t* xrow1 = hT + ((size_t)b * kN + n0 + 16 + col) * kDH;
  v8f acc0 = {}, acc1 = {};
#pragma unroll 4
  for (int k0 = 0; k0 < kDH; k0 += 32) {
    if (k0 + 64 < kDH) __builtin_prefetch(wrow + k0 + 64, 0, 3);
    v16h a = frag_row(wrow + k0, hi);
    v16h b0 = frag_row(xrow0 + k0, hi);
    acc0 = wmma_f16(a, b0, acc0);
    if (have2) {
      v16h b1 = frag_row(xrow1 + k0, hi);
      acc1 = wmma_f16(a, b1, acc1);
    }
  }
  float* ob = Out + (size_t)b * kDim * kN;
#pragma unroll
  for (int r = 0; r < 8; ++r) {
    int o = o0 + r + 8 * hi;
    float bo = bias[o];
    ob[(size_t)o * kN + n0 + col] = hswish(acc0[r] + bo);
    if (have2) ob[(size_t)o * kN + n0 + 16 + col] = hswish(acc1[r] + bo);
  }
}

// ---------------- host launcher ----------------
extern "C" void kernel_launch(void* const* d_in, const int* in_sizes, int n_in,
                              void* d_out, int out_size, void* d_ws, size_t ws_size,
                              hipStream_t stream) {
  const float* x   = (const float*)d_in[0];
  const float* qw  = (const float*)d_in[1];
  const float* qg  = (const float*)d_in[2];
  const float* qb  = (const float*)d_in[3];
  const float* qm  = (const float*)d_in[4];
  const float* qv  = (const float*)d_in[5];
  const float* ab  = (const float*)d_in[6];
  const float* pw  = (const float*)d_in[7];
  const float* pg  = (const float*)d_in[8];
  const float* pb  = (const float*)d_in[9];
  const float* pm  = (const float*)d_in[10];
  const float* pv  = (const float*)d_in[11];
  const int*   idx = (const int*)d_in[12];
  float* out = (float*)d_out;
  const int n_off = in_sizes[6] / kHeads;

  // workspace carve-up (256B aligned)
  char* ws = (char*)d_ws;
  size_t off = 0;
  auto take = [&](size_t bytes) {
    char* p = ws + off;
    off = (off + bytes + 255) & ~(size_t)255;
    return p;
  };
  half_t* qwh   = (half_t*)take((size_t)kHQKV * kDim * sizeof(half_t));
  float*  qbias = (float*) take((size_t)kHQKV * sizeof(float));
  half_t* pwh   = (half_t*)take((size_t)kDim * kDH * sizeof(half_t));
  float*  pbias = (float*) take((size_t)kDim * sizeof(float));
  half_t* xT    = (half_t*)take((size_t)kBatch * kN * kDim * sizeof(half_t));
  half_t* qT    = (half_t*)take((size_t)kBatch * kHeads * kN * kKD * sizeof(half_t));
  half_t* kT    = (half_t*)take((size_t)kBatch * kHeads * kN * kKD * sizeof(half_t));
  half_t* vbuf  = (half_t*)take((size_t)kBatch * kHeads * kD * kN * sizeof(half_t));
  float*  biasF = (float*) take((size_t)kHeads * kN * kN * sizeof(float));
  half_t* hT    = (half_t*)take((size_t)kBatch * kN * kDH * sizeof(half_t));
  (void)ws_size; (void)n_in; (void)out_size;

  prep_weights<<<1024, 256, 0, stream>>>(qw, qg, qb, qm, qv, pw, pg, pb, pm, pv,
                                         qwh, qbias, pwh, pbias);
  prep_x_bias<<<4096, 256, 0, stream>>>(x, xT, ab, idx, biasF, n_off);

  qkv_gemm<<<dim3(kHQKV / 256, kN / 16, kBatch), 256, 0, stream>>>(
      qwh, qbias, xT, qT, kT, vbuf);
  attn_kernel<<<dim3(13, kHeads, kBatch), 128, 0, stream>>>(qT, kT, vbuf, biasF,
                                                            hT);
  proj_gemm<<<dim3(kDim / 128, 25, kBatch), 256, 0, stream>>>(pwh, pbias, hT,
                                                              out);
}
